// BaselineSAGELayer_3229815407098
// MI455X (gfx1250) — matrix operand order repaired and checked by reading, not compiled, not run on hardware
//
#include <hip/hip_runtime.h>
#include <hip/hip_bf16.h>

// ---------------------------------------------------------------------------
// GraphSAGE layer for MI455X (gfx1250).
//  Pass 1: zero accumulators
//  Pass 2: edge scatter (1 wave / edge, float4 gather + 4 global f32 atomics/lane)
//  Pass 3: build A = [mean | x] in f16 (row-major [N][256])
//  Pass 4: pack B = [W_l;W_r]^T into per-lane WMMA fragment order (f16)
//  Pass 5: WMMA GEMM, 80 rows/block (5 M-subtiles per wave, B-fragment reuse),
//          v_wmma_f32_16x16x32_f16 + bias + fused L2 row-normalize
// ---------------------------------------------------------------------------

#define N_NODES 50000
#define IN_C    128
#define OUT_C   128
#define N_EDGES 800000
#define K_TOT   256          // [mean | x] concatenated along K
#define MT      5            // 16-row M-subtiles per block (80 rows; 50000 = 625*80)
#define MROWS   (MT * 16)

typedef __attribute__((ext_vector_type(16))) _Float16 v16h;
typedef __attribute__((ext_vector_type(8)))  _Float16 v8h;
typedef __attribute__((ext_vector_type(8)))  float    v8f;

// workspace byte offsets
#define AGG_OFF  ((size_t)0)
#define CNT_OFF  ((size_t)N_NODES * IN_C * 4)                 // 25,600,000
#define A_OFF    (CNT_OFF + (size_t)N_NODES * 4)              // 25,800,000 (16B aligned)
#define B_OFF    (A_OFF + (size_t)N_NODES * K_TOT * 2)        // 51,400,000 (32B aligned)

__global__ void sage_zero(float* __restrict__ p, int n) {
    int i = blockIdx.x * blockDim.x + threadIdx.x;
    if (i < n) p[i] = 0.0f;
}

// one wave32 per edge: lane l gathers x[src][4l..4l+3], atomically adds into agg[dst]
__global__ void sage_edge_scatter(const float* __restrict__ x,
                                  const int* __restrict__ ei,
                                  float* __restrict__ agg,
                                  float* __restrict__ cnt) {
    int wid  = (blockIdx.x << 3) + (threadIdx.x >> 5);   // 8 waves / block
    int lane = threadIdx.x & 31;
    if (wid >= N_EDGES) return;
    int src = ei[wid];             // row 0 of edge_index
    int dst = ei[N_EDGES + wid];   // row 1 of edge_index
    const float4 v = *(const float4*)(x + (size_t)src * IN_C + lane * 4);
    float* a = agg + (size_t)dst * IN_C + lane * 4;
    __hip_atomic_fetch_add(a + 0, v.x, __ATOMIC_RELAXED, __HIP_MEMORY_SCOPE_AGENT);
    __hip_atomic_fetch_add(a + 1, v.y, __ATOMIC_RELAXED, __HIP_MEMORY_SCOPE_AGENT);
    __hip_atomic_fetch_add(a + 2, v.z, __ATOMIC_RELAXED, __HIP_MEMORY_SCOPE_AGENT);
    __hip_atomic_fetch_add(a + 3, v.w, __ATOMIC_RELAXED, __HIP_MEMORY_SCOPE_AGENT);
    if (lane == 0)
        __hip_atomic_fetch_add(cnt + dst, 1.0f, __ATOMIC_RELAXED, __HIP_MEMORY_SCOPE_AGENT);
}

// A[node][k]    = mean[node][k]  (k <  128)
// A[node][128+k]= x[node][k]
__global__ void sage_build_a(const float* __restrict__ x,
                             const float* __restrict__ agg,
                             const float* __restrict__ cnt,
                             _Float16* __restrict__ A) {
    int gid = blockIdx.x * blockDim.x + threadIdx.x;   // N_NODES*IN_C threads
    if (gid >= N_NODES * IN_C) return;
    int node = gid >> 7;
    int k    = gid & 127;
    float inv  = 1.0f / fmaxf(cnt[node], 1.0f);
    float mean = agg[gid] * inv;
    _Float16* row = A + (size_t)node * K_TOT;
    row[k]         = (_Float16)mean;
    row[IN_C + k]  = (_Float16)x[gid];
}

// Pack B (256x128, B[k][n] = k<128 ? W_l[n][k] : W_r[n][k-128]) into a
// per-lane B-fragment order, contiguous K per lane half (same order as the A loads):
//   Bp[((t*8+kk)*32 + lid)*16 + e] = B[kk*32 + (lid>>4)*16 + e][t*16 + (lid&15)]
// so in the GEMM each lane does one contiguous 32B load per K-step.
__global__ void sage_pack_b(const float* __restrict__ W_l,
                            const float* __restrict__ W_r,
                            _Float16* __restrict__ Bp) {
    int gid = blockIdx.x * blockDim.x + threadIdx.x;   // 8*8*32*16 = 32768 threads
    if (gid >= 8 * 8 * 32 * 16) return;
    int e   = gid & 15;
    int lid = (gid >> 4) & 31;
    int kk  = (gid >> 9) & 7;
    int t   = gid >> 12;
    int kg  = kk * 32 + ((lid >> 4) << 4) + e;         // global K index 0..255
    int col = t * 16 + (lid & 15);                     // output channel 0..127
    float v = (kg < IN_C) ? W_l[(size_t)col * IN_C + kg]
                          : W_r[(size_t)col * IN_C + (kg - IN_C)];
    Bp[gid] = (_Float16)v;
}

// 80-row tile per block; 8 waves, one 16-col tile each, 5 M-subtiles per wave.
// Per K-step: 1 B fragment reused across 5 WMMAs (5 A fragments from LDS).
// Epilogue: + bias, fused L2 row-normalize via shfl butterfly + LDS reduction.
__global__ __launch_bounds__(256) void sage_gemm(const _Float16* __restrict__ A,
                                                 const _Float16* __restrict__ Bp,
                                                 const float* __restrict__ bias,
                                                 float* __restrict__ out) {
    __shared__ _Float16 As[MROWS * K_TOT];   // 40 KB
    __shared__ float    psum[MROWS][8];      // 2.5 KB

    const int tid = threadIdx.x;
    const int w   = tid >> 5;
    const int lid = tid & 31;
    const int m0  = blockIdx.x * MROWS;

    // stage contiguous 80x256 f16 A-tile into LDS (b128 copies, 10 per thread)
    const uint4* gsrc = (const uint4*)(A + (size_t)m0 * K_TOT);
    uint4* lds4 = (uint4*)As;
#pragma unroll
    for (int i = 0; i < 2 * MT; ++i)
        lds4[tid + 256 * i] = gsrc[tid + 256 * i];
    __syncthreads();

    const int row = lid & 15;   // M within subtile (A frag) / N within tile (B frag)
    const int hi  = lid >> 4;

    v8f c[MT];
#pragma unroll
    for (int t = 0; t < MT; ++t)
        c[t] = (v8f){0.f, 0.f, 0.f, 0.f, 0.f, 0.f, 0.f, 0.f};

    const _Float16* bbase = Bp + ((size_t)w * 8 * 32 + lid) * 16;

#pragma unroll
    for (int kk = 0; kk < 8; ++kk) {
        // B fragment: one contiguous 32B per lane (pre-packed); reused 5x
        v16h b = *(const v16h*)(bbase + (size_t)kk * 32 * 16);
#pragma unroll
        for (int t = 0; t < MT; ++t) {
            // A fragment: lanes 0-15 hold K kk*32+{0..15}; lanes 16-31 +16
            union { v16h v; v8h h[2]; } a;
            const _Float16* ap = As + (t * 16 + row) * K_TOT + kk * 32 + hi * 16;
            a.h[0] = *(const v8h*)ap;          // 16B-aligned ds_load_b128
            a.h[1] = *(const v8h*)(ap + 8);
            c[t] = __builtin_amdgcn_wmma_f32_16x16x32_f16(
                       false, a.v, false, b, (short)0, c[t], false, false);
        }
    }

    const int   col = w * 16 + row;        // output channel
    const float bv  = bias[col];

    // bias + per-row sum of squares (16 cols reduced within each half-wave)
#pragma unroll
    for (int t = 0; t < MT; ++t) {
#pragma unroll
        for (int r = 0; r < 8; ++r) {
            float v = c[t][r] + bv;
            c[t][r] = v;
            float s = v * v;
            s += __shfl_xor(s, 1, 16);
            s += __shfl_xor(s, 2, 16);
            s += __shfl_xor(s, 4, 16);
            s += __shfl_xor(s, 8, 16);
            if (row == 0) psum[t * 16 + r + 8 * hi][w] = s;   // lanes 0 & 16 publish
        }
    }
    __syncthreads();

#pragma unroll
    for (int t = 0; t < MT; ++t) {
#pragma unroll
        for (int r = 0; r < 8; ++r) {
            const int orow = t * 16 + r + 8 * hi;
            float tot = 0.f;
#pragma unroll
            for (int w2 = 0; w2 < 8; ++w2) tot += psum[orow][w2];
            float nrm = fmaxf(sqrtf(tot), 1e-12f);
            out[(size_t)(m0 + orow) * OUT_C + col] = c[t][r] / nrm;
        }
    }
}

extern "C" void kernel_launch(void* const* d_in, const int* in_sizes, int n_in,
                              void* d_out, int out_size, void* d_ws, size_t ws_size,
                              hipStream_t stream) {
    const float* x   = (const float*)d_in[0];
    const int*   ei  = (const int*)d_in[1];
    const float* W_l = (const float*)d_in[2];
    const float* b_l = (const float*)d_in[3];
    const float* W_r = (const float*)d_in[4];
    float*       out = (float*)d_out;

    char* ws = (char*)d_ws;
    float*    agg = (float*)(ws + AGG_OFF);
    float*    cnt = (float*)(ws + CNT_OFF);
    _Float16* Af  = (_Float16*)(ws + A_OFF);
    _Float16* Bp  = (_Float16*)(ws + B_OFF);

    const int nzero = N_NODES * IN_C + N_NODES;
    sage_zero<<<(nzero + 255) / 256, 256, 0, stream>>>(agg, nzero);

    sage_edge_scatter<<<N_EDGES / 8, 256, 0, stream>>>(x, ei, agg, cnt);

    sage_build_a<<<(N_NODES * IN_C + 255) / 256, 256, 0, stream>>>(x, agg, cnt, Af);

    sage_pack_b<<<(8 * 8 * 32 * 16) / 256, 256, 0, stream>>>(W_l, W_r, Bp);

    sage_gemm<<<N_NODES / MROWS, 256, 0, stream>>>(Af, Bp, b_l, out);
}
